// Linear_7378753814899
// MI455X (gfx1250) — 8-bit matrix operand order repaired and checked by reading, compiled, not run on hardware
//
#include <hip/hip_runtime.h>

typedef int   v4i  __attribute__((ext_vector_type(4)));
typedef int   v16i __attribute__((ext_vector_type(16)));
typedef float v8f  __attribute__((ext_vector_type(8)));

#define M_DIM 8192
#define K_DIM 4096
#define N_DIM 4096
#define KBLK  128
#define KB    (K_DIM / KBLK)
#define LDT   144

#if defined(__gfx1250__) && __has_builtin(__builtin_amdgcn_global_load_async_to_lds_b128)
#define HAVE_ASYNC_LDS 1
#else
#define HAVE_ASYNC_LDS 0
#endif

#if HAVE_ASYNC_LDS
typedef __attribute__((address_space(1))) v4i* gv4p;
typedef __attribute__((address_space(3))) v4i* lv4p;

__device__ __forceinline__ void async_copy16(const unsigned char* gsrc, unsigned char* ldst) {
  __builtin_amdgcn_global_load_async_to_lds_b128(
      (gv4p)(uintptr_t)gsrc,
      (lv4p)(unsigned int)(uintptr_t)ldst, 0, 0);
}
__device__ __forceinline__ void wait_async0() {
  asm volatile("s_wait_asynccnt 0" ::: "memory");
}
#endif

__device__ __forceinline__ unsigned int f32_to_e4m3(float x) {
  unsigned int sgn = (__float_as_uint(x) >> 31) << 7;
  float a = fminf(fabsf(x), 448.0f);
  if (a < 0.015625f) {
    unsigned int m = (unsigned int)rintf(a * 512.0f);
    return sgn | m;
  }
  unsigned int au = __float_as_uint(a);
  unsigned int r  = au + 0x7FFFFu + ((au >> 20) & 1u);
  unsigned int e  = (r >> 23) - 127u + 7u;
  return sgn | (e << 3) | ((r >> 20) & 7u);
}

__global__ __launch_bounds__(256) void quant_x_kernel(
    const float* __restrict__ x, unsigned char* __restrict__ xq,
    float* __restrict__ sx) {
  const int wid  = blockIdx.x * 8 + (threadIdx.x >> 5);
  const int lane = threadIdx.x & 31;
  const int m  = wid / KB;
  const int kb = wid % KB;

  const float4 v = *(const float4*)(x + (size_t)m * K_DIM + kb * KBLK + lane * 4);
  float am = fmaxf(fmaxf(fabsf(v.x), fabsf(v.y)), fmaxf(fabsf(v.z), fabsf(v.w)));
#pragma unroll
  for (int off = 16; off >= 1; off >>= 1)
    am = fmaxf(am, __shfl_xor(am, off, 32));

  const float s   = (am == 0.0f) ? 1.0f : am * (1.0f / 448.0f);
  const float inv = 1.0f / s;
  const unsigned int p = f32_to_e4m3(v.x * inv)
                       | (f32_to_e4m3(v.y * inv) << 8)
                       | (f32_to_e4m3(v.z * inv) << 16)
                       | (f32_to_e4m3(v.w * inv) << 24);
  *(unsigned int*)(xq + (size_t)m * K_DIM + kb * KBLK + lane * 4) = p;
  if (lane == 0) sx[m * KB + kb] = s;
}

__global__ __launch_bounds__(256) void quant_w_kernel(
    const float* __restrict__ w, unsigned char* __restrict__ wb) {
  const size_t i = (size_t)blockIdx.x * 256 + threadIdx.x;
  const float4 v = ((const float4*)w)[i];
  ((unsigned int*)wb)[i] = f32_to_e4m3(v.x)
                         | (f32_to_e4m3(v.y) << 8)
                         | (f32_to_e4m3(v.z) << 16)
                         | (f32_to_e4m3(v.w) << 24);
}

union AFrag { v16i v; unsigned long long d[8]; };
union BFrag { v16i v; uint4 q[4]; };

__global__ __launch_bounds__(256) void fp8_gemm_kernel(
    const unsigned char* __restrict__ Aq,
    const unsigned char* __restrict__ Bq,
    const float* __restrict__ sx,
    const float* __restrict__ sw,
    float* __restrict__ out)
{
  __shared__ __align__(16) unsigned char As[2][128 * LDT];
  __shared__ __align__(16) unsigned char Bs[2][128 * LDT];
  __shared__ float sxs[128 * KB];

  const int nbBlk = blockIdx.x;
  const int mbBlk = blockIdx.y;
  const int tid  = threadIdx.x;
  const int lane = tid & 31;
  const int wave = tid >> 5;
  const int wm   = wave >> 1;
  const int wn   = wave & 1;
  const int l16  = lane & 15;
  const int half = lane >> 4;

  const int Mbase = mbBlk * 128;
  const int Nbase = nbBlk * 128;

  const int srow = tid >> 3;
  const int scol = (tid & 7) * 16;

  const float swv = sw[nbBlk * KB + (tid & 31)];
#pragma unroll
  for (int j = 0; j < 16; ++j)
    sxs[tid + 256 * j] = sx[Mbase * KB + tid + 256 * j] * swv;

  float acc[2][4][8];
#pragma unroll
  for (int t = 0; t < 2; ++t)
#pragma unroll
    for (int u = 0; u < 4; ++u)
#pragma unroll
      for (int i = 0; i < 8; ++i) acc[t][u][i] = 0.0f;

#if HAVE_ASYNC_LDS
  {
#pragma unroll
    for (int r = 0; r < 4; ++r) {
      async_copy16(Aq + (size_t)(Mbase + r * 32 + srow) * K_DIM + scol,
                   &As[0][(r * 32 + srow) * LDT + scol]);
      async_copy16(Bq + (size_t)(Nbase + r * 32 + srow) * K_DIM + scol,
                   &Bs[0][(r * 32 + srow) * LDT + scol]);
    }
    wait_async0();
  }
#else
  {
    uint4 ra[4], rb[4];
#pragma unroll
    for (int r = 0; r < 4; ++r) {
      ra[r] = *(const uint4*)(Aq + (size_t)(Mbase + r * 32 + srow) * K_DIM + scol);
      rb[r] = *(const uint4*)(Bq + (size_t)(Nbase + r * 32 + srow) * K_DIM + scol);
    }
#pragma unroll
    for (int r = 0; r < 4; ++r) {
      *(uint4*)&As[0][(r * 32 + srow) * LDT + scol] = ra[r];
      *(uint4*)&Bs[0][(r * 32 + srow) * LDT + scol] = rb[r];
    }
  }
#endif
  __syncthreads();

  for (int kb = 0; kb < KB; ++kb) {
    const int  cur = kb & 1;
    const bool pre = (kb + 1) < KB;

#if HAVE_ASYNC_LDS
    if (pre) {
      const int nxt = cur ^ 1;
      const size_t koff = (size_t)(kb + 1) * KBLK;
#pragma unroll
      for (int r = 0; r < 4; ++r) {
        async_copy16(Aq + (size_t)(Mbase + r * 32 + srow) * K_DIM + koff + scol,
                     &As[nxt][(r * 32 + srow) * LDT + scol]);
        async_copy16(Bq + (size_t)(Nbase + r * 32 + srow) * K_DIM + koff + scol,
                     &Bs[nxt][(r * 32 + srow) * LDT + scol]);
      }
    }
#else
    uint4 ra[4], rb[4];
    if (pre) {
      const size_t koff = (size_t)(kb + 1) * KBLK;
#pragma unroll
      for (int r = 0; r < 4; ++r) {
        ra[r] = *(const uint4*)(Aq + (size_t)(Mbase + r * 32 + srow) * K_DIM + koff + scol);
        rb[r] = *(const uint4*)(Bq + (size_t)(Nbase + r * 32 + srow) * K_DIM + koff + scol);
      }
    }
#endif

    AFrag a[2];
#pragma unroll
    for (int t = 0; t < 2; ++t) {
      const unsigned char* p = &As[cur][(wm * 32 + t * 16 + l16) * LDT + half * 16];
#pragma unroll
      for (int g = 0; g < 8; ++g)
        a[t].d[g] = *(const unsigned long long*)(p + (g >> 1) * 32 + (g & 1) * 8);
    }
    BFrag b[4];
#pragma unroll
    for (int u = 0; u < 4; ++u) {
      const unsigned char* p = &Bs[cur][(wn * 64 + u * 16 + l16) * LDT + half * 16];
#pragma unroll
      for (int q = 0; q < 4; ++q)
        b[u].q[q] = *(const uint4*)(p + q * 32);
    }

    float ssc[2][8];
#pragma unroll
    for (int t = 0; t < 2; ++t)
#pragma unroll
      for (int i = 0; i < 8; ++i)
        ssc[t][i] = sxs[(wm * 32 + t * 16 + i + 8 * half) * KB + kb];

    const v8f cz = {0.f, 0.f, 0.f, 0.f, 0.f, 0.f, 0.f, 0.f};
#pragma unroll
    for (int t = 0; t < 2; ++t)
#pragma unroll
      for (int u = 0; u < 4; ++u) {
        v8f d = __builtin_amdgcn_wmma_f32_16x16x128_fp8_fp8(
            a[t].v, b[u].v, (short)0, cz, false, false);
#pragma unroll
        for (int i = 0; i < 8; ++i)
          acc[t][u][i] = fmaf(d[i], ssc[t][i], acc[t][u][i]);
      }

#if HAVE_ASYNC_LDS
    if (pre) wait_async0();
#else
    if (pre) {
      const int nxt = cur ^ 1;
#pragma unroll
      for (int r = 0; r < 4; ++r) {
        *(uint4*)&As[nxt][(r * 32 + srow) * LDT + scol] = ra[r];
        *(uint4*)&Bs[nxt][(r * 32 + srow) * LDT + scol] = rb[r];
      }
    }
#endif
    __syncthreads();
  }

#pragma unroll
  for (int t = 0; t < 2; ++t)
#pragma unroll
    for (int u = 0; u < 4; ++u) {
      const int col = Nbase + wn * 64 + u * 16 + l16;
#pragma unroll
      for (int i = 0; i < 8; ++i) {
        const int row = Mbase + wm * 32 + t * 16 + i + 8 * half;
        out[(size_t)row * N_DIM + col] = acc[t][u][i];
      }
    }
}

extern "C" void kernel_launch(void* const* d_in, const int* in_sizes, int n_in,
                              void* d_out, int out_size, void* d_ws, size_t ws_size,
                              hipStream_t stream) {
  const float* x  = (const float*)d_in[0];
  const float* wq = (const float*)d_in[1];
  const float* sw = (const float*)d_in[2];
  float* out = (float*)d_out;

  unsigned char* ws  = (unsigned char*)d_ws;
  unsigned char* xqb = ws;
  unsigned char* wqb = ws + (size_t)M_DIM * K_DIM;
  float* sx = (float*)(ws + (size_t)M_DIM * K_DIM + (size_t)N_DIM * K_DIM);

  quant_x_kernel<<<(M_DIM * KB) / 8, 256, 0, stream>>>(x, xqb, sx);
  quant_w_kernel<<<(N_DIM / 4) * (K_DIM / 256), 256, 0, stream>>>(wq, wqb);

  dim3 grid(N_DIM / 128, M_DIM / 128);
  fp8_gemm_kernel<<<grid, 256, 0, stream>>>(xqb, wqb, sx, sw, out);
}
